// GCNWithAttention_45801531244884
// MI455X (gfx1250) — matrix operand order repaired and checked by reading, not compiled, not run on hardware
//
#include <hip/hip_runtime.h>
#include <cmath>

#define IN_C   128
#define HID    64
#define OUTC   32
#define NHEADS 2
#define NEG_SLOPE 0.2f

typedef __attribute__((ext_vector_type(16))) __bf16 v16bf;
typedef __attribute__((ext_vector_type(4)))  __bf16 v4bf;
typedef __attribute__((ext_vector_type(8)))  float  v8f;

union FragC { v8f v; float e[8]; };
union Quad  { v4bf v; __bf16 e[4]; };

__device__ __forceinline__ float leaky(float x) { return x > 0.f ? x : NEG_SLOPE * x; }

// monotone float <-> uint key mapping so atomicMax(u32) == float max
__device__ __forceinline__ unsigned f2key(float f) {
  unsigned b = __float_as_uint(f);
  return (b & 0x80000000u) ? ~b : (b | 0x80000000u);
}
__device__ __forceinline__ float key2f(unsigned k) {
  return (k & 0x80000000u) ? __uint_as_float(k & 0x7fffffffu) : __uint_as_float(~k);
}

// ---------------- utility kernels ----------------
__global__ void k_fill(float* __restrict__ p, float v, long long n) {
  long long g = (long long)blockIdx.x * blockDim.x + threadIdx.x;
  if (g < n) p[g] = v;
}

__global__ void k_deg(const long long* __restrict__ dst, float* __restrict__ deg, long long E) {
  long long g = (long long)blockIdx.x * blockDim.x + threadIdx.x;
  if (g < E) atomicAdd(&deg[dst[g]], 1.0f);
}

__global__ void k_dinv(const float* __restrict__ deg, float* __restrict__ dinv, long long n) {
  long long g = (long long)blockIdx.x * blockDim.x + threadIdx.x;
  if (g < n) dinv[g] = rsqrtf(deg[g]);   // deg >= 1 (self loop)
}

// ---------------- WMMA GEMM ----------------
// C[n,M] = A[n,K] x B[K,M], fp32 accuracy via bf16 hi/lo split (3 WMMAs).
// Block = 128 threads = 4 waves; block tile 64x64; wave w owns rows w*16..w*16+15
// and all four 16-col tiles -> 12 v_wmma per K-step with A-fragment reuse.
// LDS staging is in *fragment layout*: lane's 16 bf16 elements contiguous, so
// fragment reads are 2x ds_load_b128 (no per-element gather).
__global__ __launch_bounds__(128)
void k_gemm_wmma(const float* __restrict__ A, const float* __restrict__ B,
                 float* __restrict__ C, int n, int K, int M) {
  __shared__ __align__(32) __bf16 AfH[4][32][16];
  __shared__ __align__(32) __bf16 AfL[4][32][16];
  __shared__ __align__(32) __bf16 BfH[4][32][16];
  __shared__ __align__(32) __bf16 BfL[4][32][16];

  const int t    = threadIdx.x;
  const int lane = t & 31;
  const int wv   = t >> 5;
  const int row0 = blockIdx.x * 64;
  const int col0 = blockIdx.y * 64;

  FragC acc[4];
#pragma unroll
  for (int j = 0; j < 4; ++j)
#pragma unroll
    for (int r = 0; r < 8; ++r) acc[j].e[r] = 0.0f;

  for (int k0 = 0; k0 < K; k0 += 32) {
    // ---- stage A tile (64 rows x 32 K) via float4 loads, store in fragment layout
#pragma unroll
    for (int i = t; i < 512; i += 128) {
      int r  = i >> 3;          // tile row 0..63
      int c4 = (i & 7) * 4;     // K within tile, 4-aligned
      float vv[4] = {0.f, 0.f, 0.f, 0.f};
      if (row0 + r < n) {
        float4 v = *(const float4*)&A[(long long)(row0 + r) * K + (k0 + c4)];
        vv[0] = v.x; vv[1] = v.y; vv[2] = v.z; vv[3] = v.w;
      }
      int w  = r >> 4;                              // wave row-tile
      int la = (((c4 >> 3) & 1) << 4) | (r & 15);   // dest lane (K-half select)
      int e0 = ((c4 & 16) >> 1) | (c4 & 7);         // first fragment element
      Quad hv, lv;
#pragma unroll
      for (int q = 0; q < 4; ++q) {
        __bf16 h = (__bf16)vv[q];
        hv.e[q] = h;
        lv.e[q] = (__bf16)(vv[q] - (float)h);
      }
      *(v4bf*)&AfH[w][la][e0] = hv.v;   // single ds_store_b64 each
      *(v4bf*)&AfL[w][la][e0] = lv.v;
    }
    // ---- stage B tile (32 K x 64 cols) via float4 loads
#pragma unroll
    for (int i = t; i < 512; i += 128) {
      int r  = i >> 4;          // K within tile 0..31
      int c4 = (i & 15) * 4;    // col within tile
      float4 v = *(const float4*)&B[(long long)(k0 + r) * M + (col0 + c4)];
      float vv[4] = {v.x, v.y, v.z, v.w};
      int kh = (r >> 3) & 1;
      int el = ((r & 16) >> 1) | (r & 7);
#pragma unroll
      for (int q = 0; q < 4; ++q) {
        int col = c4 + q;
        int la  = (kh << 4) | (col & 15);
        int j   = col >> 4;
        __bf16 h = (__bf16)vv[q];
        BfH[j][la][el] = h;
        BfL[j][la][el] = (__bf16)(vv[q] - (float)h);
      }
    }
    __syncthreads();

    // ---- compute: A fragment reused across 4 col tiles; 12 WMMAs per K-step
    v16bf ah = *(const v16bf*)&AfH[wv][lane][0];
    v16bf al = *(const v16bf*)&AfL[wv][lane][0];
#pragma unroll
    for (int j = 0; j < 4; ++j) {
      v16bf bh = *(const v16bf*)&BfH[j][lane][0];
      v16bf bl = *(const v16bf*)&BfL[j][lane][0];
      acc[j].v = __builtin_amdgcn_wmma_f32_16x16x32_bf16(false, ah, false, bh, (short)0, acc[j].v, false, false);
      acc[j].v = __builtin_amdgcn_wmma_f32_16x16x32_bf16(false, al, false, bh, (short)0, acc[j].v, false, false);
      acc[j].v = __builtin_amdgcn_wmma_f32_16x16x32_bf16(false, ah, false, bl, (short)0, acc[j].v, false, false);
    }
    __syncthreads();
  }

  // ---- store C: VGPR r -> row (lane>>4)*8 + r, col = lane&15 within each 16-col tile
#pragma unroll
  for (int j = 0; j < 4; ++j) {
    int col = col0 + j * 16 + (lane & 15);
#pragma unroll
    for (int r = 0; r < 8; ++r) {
      int row = row0 + wv * 16 + ((lane >> 4) << 3) + r;
      if (row < n) C[(long long)row * M + col] = acc[j].e[r];
    }
  }
}

// ---------------- GCN aggregation ----------------
template <int M>
__global__ void k_gcn_self(const float* __restrict__ h, const float* __restrict__ dinv,
                           float* __restrict__ out, long long n) {
  long long g = (long long)blockIdx.x * blockDim.x + threadIdx.x;
  if (g >= n * M) return;
  long long i = g / M;
  float di = dinv[i];
  out[g] = h[g] * di * di;   // self-loop contribution seeds the accumulator
}

template <int M>
__global__ void k_gcn_edge(const long long* __restrict__ src, const long long* __restrict__ dst,
                           const float* __restrict__ h, const float* __restrict__ dinv,
                           float* __restrict__ out, long long E) {
  long long g = (long long)blockIdx.x * blockDim.x + threadIdx.x;
  if (g >= E * M) return;
  long long e = g / M;
  int f = (int)(g - e * M);
  long long s = src[e], d = dst[e];
  float w = dinv[s] * dinv[d];
  atomicAdd(&out[d * M + f], h[s * M + f] * w);
}

template <int M>
__global__ void k_bias_relu(float* __restrict__ io, const float* __restrict__ b, long long n) {
  long long g = (long long)blockIdx.x * blockDim.x + threadIdx.x;
  if (g >= n * M) return;
  int f = (int)(g % M);
  float v = io[g] + b[f];
  io[g] = v > 0.f ? v : 0.f;
}

// ---------------- GAT ----------------
template <int H, int C>
__global__ void k_gat_scores(const float* __restrict__ hw, const float* __restrict__ atts,
                             const float* __restrict__ attd, float* __restrict__ as_,
                             float* __restrict__ ad_, unsigned* __restrict__ mx, long long n) {
  long long g = (long long)blockIdx.x * blockDim.x + threadIdx.x;
  if (g >= n * H) return;
  long long i = g / H;
  int h = (int)(g - i * H);
  const float* base = hw + i * (H * C) + h * C;
  float s = 0.f, d = 0.f;
#pragma unroll
  for (int c = 0; c < C; ++c) {
    float v = base[c];
    s += v * atts[h * C + c];
    d += v * attd[h * C + c];
  }
  as_[g] = s;
  ad_[g] = d;
  mx[g] = f2key(leaky(s + d));   // self-loop edge seeds segment_max
}

template <int H>
__global__ void k_gat_edge_max(const long long* __restrict__ src, const long long* __restrict__ dst,
                               const float* __restrict__ as_, const float* __restrict__ ad_,
                               unsigned* __restrict__ mx, long long E) {
  long long g = (long long)blockIdx.x * blockDim.x + threadIdx.x;
  if (g >= E * H) return;
  long long e = g / H;
  int h = (int)(g - e * H);
  long long s = src[e], d = dst[e];
  float ev = leaky(as_[s * H + h] + ad_[d * H + h]);
  atomicMax(&mx[d * H + h], f2key(ev));
}

template <int H, int C>
__global__ void k_gat_self(const float* __restrict__ hw, const float* __restrict__ as_,
                           const float* __restrict__ ad_, const unsigned* __restrict__ mx,
                           float* __restrict__ sm, float* __restrict__ out, long long n) {
  long long g = (long long)blockIdx.x * blockDim.x + threadIdx.x;
  if (g >= n * H * C) return;
  long long i = g / (H * C);
  int f = (int)(g - i * (H * C));
  int h = f / C, c = f - h * C;
  float m = key2f(mx[i * H + h]);
  float ex = expf(leaky(as_[i * H + h] + ad_[i * H + h]) - m);
  out[g] = hw[g] * ex;
  if (c == 0) sm[i * H + h] = ex;
}

template <int H, int C>
__global__ void k_gat_edge_agg(const long long* __restrict__ src, const long long* __restrict__ dst,
                               const float* __restrict__ hw, const float* __restrict__ as_,
                               const float* __restrict__ ad_, const unsigned* __restrict__ mx,
                               float* __restrict__ out, float* __restrict__ sm, long long E) {
  long long g = (long long)blockIdx.x * blockDim.x + threadIdx.x;
  if (g >= E * H * C) return;
  long long e = g / (H * C);
  int f = (int)(g - e * (H * C));
  int h = f / C, c = f - h * C;
  long long s = src[e], d = dst[e];
  float ev = leaky(as_[s * H + h] + ad_[d * H + h]);
  float ex = expf(ev - key2f(mx[d * H + h]));
  atomicAdd(&out[d * (H * C) + f], hw[s * (H * C) + f] * ex);
  if (c == 0) atomicAdd(&sm[d * H + h], ex);
}

template <int H, int C>
__global__ void k_gat_final_concat(float* __restrict__ io, const float* __restrict__ sm,
                                   const float* __restrict__ b, long long n) {
  long long g = (long long)blockIdx.x * blockDim.x + threadIdx.x;
  if (g >= n * H * C) return;
  long long i = g / (H * C);
  int f = (int)(g - i * (H * C));
  int h = f / C;
  float v = io[g] / sm[i * H + h] + b[f];
  io[g] = v > 0.f ? v : 0.f;
}

// mean over heads + bias + log_softmax; 32 lanes == OUTC, one wave32 per node
__global__ __launch_bounds__(128)
void k_gat2_out(const float* __restrict__ outb, const float* __restrict__ sm,
                const float* __restrict__ b, float* __restrict__ y, long long n) {
  int lane = threadIdx.x & 31;
  long long node = (long long)blockIdx.x * 4 + (threadIdx.x >> 5);
  if (node >= n) return;
  float v = 0.f;
#pragma unroll
  for (int h = 0; h < NHEADS; ++h)
    v += outb[node * (NHEADS * OUTC) + h * OUTC + lane] / sm[node * NHEADS + h];
  v = v * (1.0f / NHEADS) + b[lane];
  float m = v;
#pragma unroll
  for (int o = 16; o > 0; o >>= 1) m = fmaxf(m, __shfl_xor(m, o));
  float s = expf(v - m);
#pragma unroll
  for (int o = 16; o > 0; o >>= 1) s += __shfl_xor(s, o);
  y[node * OUTC + lane] = (v - m) - logf(s);
}

// ---------------- host launch ----------------
extern "C" void kernel_launch(void* const* d_in, const int* in_sizes, int n_in,
                              void* d_out, int out_size, void* d_ws, size_t ws_size,
                              hipStream_t stream) {
  (void)n_in; (void)out_size; (void)ws_size;

  const float*     x    = (const float*)d_in[0];
  const long long* ei   = (const long long*)d_in[1];   // int64 per reference
  const float*     W1   = (const float*)d_in[2];
  const float*     b1   = (const float*)d_in[3];
  const float*     W2   = (const float*)d_in[4];
  const float*     b2   = (const float*)d_in[5];
  const float*     Wg1  = (const float*)d_in[6];
  const float*     as1  = (const float*)d_in[7];
  const float*     ad1  = (const float*)d_in[8];
  const float*     bg1  = (const float*)d_in[9];
  const float*     Wg2  = (const float*)d_in[10];
  const float*     as2  = (const float*)d_in[11];
  const float*     ad2  = (const float*)d_in[12];
  const float*     bg2  = (const float*)d_in[13];

  const long long n = in_sizes[0] / IN_C;
  const long long E = in_sizes[1] / 2;
  const long long* src = ei;
  const long long* dst = ei + E;

  // workspace carve-out
  size_t off = 0;
  auto carve = [&](size_t bytes) -> char* {
    off = (off + 255) & ~(size_t)255;
    char* p = (char*)d_ws + off;
    off += bytes;
    return p;
  };
  float*    deg  = (float*)carve((size_t)n * 4);
  float*    dinv = (float*)carve((size_t)n * 4);
  float*    bufG = (float*)carve((size_t)n * 128 * 4);   // gemm output
  float*    bufO = (float*)carve((size_t)n * 128 * 4);   // aggregation / activation
  float*    bufH = (float*)carve((size_t)n * 128 * 4);   // aggregation / activation
  float*    asb  = (float*)carve((size_t)n * NHEADS * 4);
  float*    adb  = (float*)carve((size_t)n * NHEADS * 4);
  unsigned* mx   = (unsigned*)carve((size_t)n * NHEADS * 4);
  float*    smb  = (float*)carve((size_t)n * NHEADS * 4);

  auto nb = [](long long total, int bs) { return dim3((unsigned)((total + bs - 1) / bs)); };
  const dim3 B256(256);
  const unsigned rowBlks = (unsigned)((n + 63) / 64);

  // degree + rsqrt (shared by both GCN layers)
  k_fill<<<nb(n, 256), B256, 0, stream>>>(deg, 1.0f, n);
  k_deg<<<nb(E, 256), B256, 0, stream>>>(dst, deg, E);
  k_dinv<<<nb(n, 256), B256, 0, stream>>>(deg, dinv, n);

  // ---- GCN layer 1: x[N,128] @ W1[128,64] ----
  k_gemm_wmma<<<dim3(rowBlks, HID / 64), 128, 0, stream>>>(x, W1, bufG, (int)n, IN_C, HID);
  k_gcn_self<HID><<<nb(n * HID, 256), B256, 0, stream>>>(bufG, dinv, bufO, n);
  k_gcn_edge<HID><<<nb(E * HID, 256), B256, 0, stream>>>(src, dst, bufG, dinv, bufO, E);
  k_bias_relu<HID><<<nb(n * HID, 256), B256, 0, stream>>>(bufO, b1, n);

  // ---- GCN layer 2: h[N,64] @ W2[64,64] ----
  k_gemm_wmma<<<dim3(rowBlks, HID / 64), 128, 0, stream>>>(bufO, W2, bufG, (int)n, HID, HID);
  k_gcn_self<HID><<<nb(n * HID, 256), B256, 0, stream>>>(bufG, dinv, bufH, n);
  k_gcn_edge<HID><<<nb(E * HID, 256), B256, 0, stream>>>(src, dst, bufG, dinv, bufH, E);
  k_bias_relu<HID><<<nb(n * HID, 256), B256, 0, stream>>>(bufH, b2, n);

  // ---- GAT layer 1: h[N,64] @ Wg1[64,128], heads=2, C=64, concat ----
  k_gemm_wmma<<<dim3(rowBlks, (NHEADS * HID) / 64), 128, 0, stream>>>(bufH, Wg1, bufG, (int)n, HID, NHEADS * HID);
  k_gat_scores<NHEADS, HID><<<nb(n * NHEADS, 256), B256, 0, stream>>>(bufG, as1, ad1, asb, adb, mx, n);
  k_gat_edge_max<NHEADS><<<nb(E * NHEADS, 256), B256, 0, stream>>>(src, dst, asb, adb, mx, E);
  k_gat_self<NHEADS, HID><<<nb(n * NHEADS * HID, 256), B256, 0, stream>>>(bufG, asb, adb, mx, smb, bufO, n);
  k_gat_edge_agg<NHEADS, HID><<<nb(E * NHEADS * HID, 256), B256, 0, stream>>>(src, dst, bufG, asb, adb, mx, bufO, smb, E);
  k_gat_final_concat<NHEADS, HID><<<nb(n * NHEADS * HID, 256), B256, 0, stream>>>(bufO, smb, bg1, n);

  // ---- GAT layer 2: h[N,128] @ Wg2[128,64], heads=2, C=32, mean + log_softmax ----
  k_gemm_wmma<<<dim3(rowBlks, (NHEADS * OUTC) / 64), 128, 0, stream>>>(bufO, Wg2, bufG, (int)n, NHEADS * HID, NHEADS * OUTC);
  k_gat_scores<NHEADS, OUTC><<<nb(n * NHEADS, 256), B256, 0, stream>>>(bufG, as2, ad2, asb, adb, mx, n);
  k_gat_edge_max<NHEADS><<<nb(E * NHEADS, 256), B256, 0, stream>>>(src, dst, asb, adb, mx, E);
  k_gat_self<NHEADS, OUTC><<<nb(n * NHEADS * OUTC, 256), B256, 0, stream>>>(bufG, asb, adb, mx, smb, bufH, n);
  k_gat_edge_agg<NHEADS, OUTC><<<nb(E * NHEADS * OUTC, 256), B256, 0, stream>>>(src, dst, bufG, asb, adb, mx, bufH, smb, E);
  k_gat2_out<<<dim3((unsigned)((n + 3) / 4)), 128, 0, stream>>>(bufH, smb, bg2, (float*)d_out, n);
}
